// MambaLayers_68564857913716
// MI455X (gfx1250) — matrix operand order repaired and checked by reading, not compiled, not run on hardware
//
#include <hip/hip_runtime.h>
#include <hip/hip_bf16.h>

#define N_LAYERS 4
#define D_MODEL  768
#define D_INNER  1536
#define D_STATE  16
#define D_CONV   4
#define DT_RANK  48
#define BATCH    2
#define SEQLEN   1024
#define MROWS    (BATCH * SEQLEN)        // 2048
#define XDBL_N   (DT_RANK + 2 * D_STATE) // 80
#define DT_PAD   64                      // dt rank padded to WMMA K granularity

typedef __attribute__((ext_vector_type(16))) __bf16 v16bf;
typedef __attribute__((ext_vector_type(8)))  float  v8f;

#define EPI_NONE          0
#define EPI_BIAS_SOFTPLUS 1

// ---------------------------------------------------------------------------
// NT GEMM via CDNA5 WMMA: C[M,N] = A[M,K] * W[N,K]^T (+ epilogue).
// fp32 operands converted to bf16 in registers, f32 accumulate via
// v_wmma_f32_16x16x32_bf16. Each wave computes a 32x16 tile: two
// accumulators share one B fragment (2 WMMAs per 32-wide K step).
// Requires: M % 32 == 0, N % 16 == 0, K % 32 == 0 (enforced by host).
// ---------------------------------------------------------------------------
__global__ __launch_bounds__(256)
void mamba_gemm_wmma_bf16(const float* __restrict__ A, int lda,
                          const float* __restrict__ W, int ldw,
                          float* __restrict__ C, int ldc,
                          const float* __restrict__ bias,
                          int M, int N, int K, int epi)
{
    const int lane = threadIdx.x & 31;
    const int wave = threadIdx.x >> 5;

    const int ntn  = N >> 4;   // 16-wide N tiles
    const int ntm  = M >> 5;   // 32-high M supertiles
    const int tile = blockIdx.x * 8 + wave;
    if (tile >= ntm * ntn) return;

    const int mt = tile / ntn;
    const int nt = tile % ntn;
    const int m0 = mt << 5;
    const int n0 = nt << 4;

    // A fragment (16-bit A 16x32 layout): lane l -> row m0+(l&15), half h=l>>4
    // holds K-chunks [h*8, h*8+8) and [16+h*8, 16+h*8+8) of each 32-K slab.
    const int  half  = lane >> 4;
    const int  kba   = half << 3;   // 0 or 8
    const int  kbw   = half << 4;   // 0 or 16
    const long arow0 = (long)(m0 + (lane & 15)) * lda;
    const long arow1 = arow0 + (long)16 * lda;
    // B fragment (16-bit B 32x16): lane l -> col n0+(l&15), K-chunks [h*8, h*8+8) and [16+h*8, 16+h*8+8)
    const long wrow  = (long)(n0 + (lane & 15)) * ldw;

    v8f acc0 = {}, acc1 = {};

    for (int k0 = 0; k0 < K; k0 += 32) {
        const float* pa0 = A + arow0 + k0 + kba;
        const float* pa1 = A + arow1 + k0 + kba;
        const float* pw  = W + wrow  + k0 + kba;

        float a0[16], a1[16], bw[16];
        *(float4*)(a0 + 0)  = *(const float4*)(pa0 + 0);
        *(float4*)(a0 + 4)  = *(const float4*)(pa0 + 4);
        *(float4*)(a0 + 8)  = *(const float4*)(pa0 + 16);
        *(float4*)(a0 + 12) = *(const float4*)(pa0 + 20);
        *(float4*)(a1 + 0)  = *(const float4*)(pa1 + 0);
        *(float4*)(a1 + 4)  = *(const float4*)(pa1 + 4);
        *(float4*)(a1 + 8)  = *(const float4*)(pa1 + 16);
        *(float4*)(a1 + 12) = *(const float4*)(pa1 + 20);
        *(float4*)(bw + 0)  = *(const float4*)(pw + 0);
        *(float4*)(bw + 4)  = *(const float4*)(pw + 4);
        *(float4*)(bw + 8)  = *(const float4*)(pw + 16);
        *(float4*)(bw + 12) = *(const float4*)(pw + 20);

        if (k0 + 32 < K) {
            __builtin_prefetch(pa0 + 32, 0, 1);
            __builtin_prefetch(pa1 + 32, 0, 1);
            __builtin_prefetch(pw + 32, 0, 1);
        }

        v16bf af0, af1, bf;
#pragma unroll
        for (int j = 0; j < 16; ++j) {
            af0[j] = (__bf16)a0[j];
            af1[j] = (__bf16)a1[j];
            bf[j]  = (__bf16)bw[j];
        }

        // (neg_a, A, neg_b, B, c_mod, C, reuse_a, reuse_b)
        acc0 = __builtin_amdgcn_wmma_f32_16x16x32_bf16(
            false, af0, false, bf, (short)0, acc0, false, false);
        acc1 = __builtin_amdgcn_wmma_f32_16x16x32_bf16(
            false, af1, false, bf, (short)0, acc1, false, false);
    }

    // C/D layout: VGPR r, lane l -> row m0 + r + 8*half, col n0 + (l&15)
    const int   n  = n0 + (lane & 15);
    const float bv = (epi == EPI_BIAS_SOFTPLUS) ? bias[n] : 0.0f;
#pragma unroll
    for (int r = 0; r < 8; ++r) {
        const int m = m0 + r + 8 * half;
        float v0 = acc0[r];
        float v1 = acc1[r];
        if (epi == EPI_BIAS_SOFTPLUS) {
            v0 += bv;
            v1 += bv;
            v0 = (v0 > 20.0f) ? v0 : __logf(1.0f + __expf(v0));
            v1 = (v1 > 20.0f) ? v1 : __logf(1.0f + __expf(v1));
        }
        C[(long)m * ldc + n]        = v0;
        C[(long)(m + 16) * ldc + n] = v1;
    }
}

// ---------------------------------------------------------------------------
// Pad dt activations: dt64[r, c] = (c < 48) ? xdbl[r, c] : 0   [2048 x 64]
// ---------------------------------------------------------------------------
__global__ __launch_bounds__(256)
void mamba_pad_dt(const float* __restrict__ xdbl, float* __restrict__ dt64)
{
    const int idx = blockIdx.x * 256 + threadIdx.x;
    if (idx >= MROWS * DT_PAD) return;
    const int c = idx & (DT_PAD - 1);
    const int r = idx >> 6;
    dt64[idx] = (c < DT_RANK) ? xdbl[(long)r * XDBL_N + c] : 0.0f;
}

// ---------------------------------------------------------------------------
// Pad dt_proj weights for all layers: dpw64[r, c] = (c<48) ? dpw[r, c] : 0
// r spans N_LAYERS * D_INNER rows.
// ---------------------------------------------------------------------------
__global__ __launch_bounds__(256)
void mamba_pad_dpw(const float* __restrict__ dpw, float* __restrict__ dpw64)
{
    const int idx = blockIdx.x * 256 + threadIdx.x;
    if (idx >= N_LAYERS * D_INNER * DT_PAD) return;
    const int c = idx & (DT_PAD - 1);
    const int r = idx >> 6;
    dpw64[idx] = (c < DT_RANK) ? dpw[(long)r * DT_RANK + c] : 0.0f;
}

// ---------------------------------------------------------------------------
// Causal depthwise conv (width 4) over time + bias + SiLU.
// Input: u_raw = xz[:, :, 0:D_INNER] (row stride 2*D_INNER). Output: u.
// ---------------------------------------------------------------------------
__global__ __launch_bounds__(256)
void mamba_conv_silu(const float* __restrict__ xz,
                     const float* __restrict__ cw,
                     const float* __restrict__ cb,
                     float* __restrict__ u)
{
    const int idx = blockIdx.x * blockDim.x + threadIdx.x;
    if (idx >= MROWS * D_INNER) return;
    const int d = idx % D_INNER;
    const int l = (idx / D_INNER) % SEQLEN;
    const int b = idx / (D_INNER * SEQLEN);

    const float* src = xz + (long)b * SEQLEN * (2 * D_INNER) + d;
    float acc = cb[d];
#pragma unroll
    for (int j = 0; j < D_CONV; ++j) {
        int t = l - (D_CONV - 1) + j;
        if (t >= 0) acc += src[(long)t * (2 * D_INNER)] * cw[d * D_CONV + j];
    }
    u[idx] = acc / (1.0f + __expf(-acc));  // silu
}

// ---------------------------------------------------------------------------
// Selective scan, fused with skip (u*D) and output gate y *= silu(z).
// One thread per (b, d): h[16] and A-row in VGPRs, B/C staged in LDS per step.
// ---------------------------------------------------------------------------
__global__ __launch_bounds__(256)
void mamba_scan(const float* __restrict__ delta,
                const float* __restrict__ u,
                const float* __restrict__ xdbl,
                const float* __restrict__ xz,
                const float* __restrict__ A_log,
                const float* __restrict__ D_param,
                float* __restrict__ y)
{
    __shared__ float sBC[2 * D_STATE];

    const int dblk = blockIdx.x % (D_INNER / 256);
    const int b    = blockIdx.x / (D_INNER / 256);
    const int d    = dblk * 256 + threadIdx.x;

    float Arow[D_STATE], h[D_STATE];
#pragma unroll
    for (int n = 0; n < D_STATE; ++n) {
        Arow[n] = -__expf(A_log[d * D_STATE + n]);
        h[n] = 0.0f;
    }
    const float dp = D_param[d];

    const float* dptr   = delta + (long)b * SEQLEN * D_INNER + d;
    const float* uptr   = u     + (long)b * SEQLEN * D_INNER + d;
    const float* zptr   = xz    + (long)b * SEQLEN * (2 * D_INNER) + D_INNER + d;
    const float* bcbase = xdbl  + (long)b * SEQLEN * XDBL_N + DT_RANK;
    float*       yptr   = y     + (long)b * SEQLEN * D_INNER + d;

    for (int l = 0; l < SEQLEN; ++l) {
        __syncthreads();
        if (threadIdx.x < 2 * D_STATE)
            sBC[threadIdx.x] = bcbase[(long)l * XDBL_N + threadIdx.x];
        __syncthreads();

        const float dv = dptr[(long)l * D_INNER];
        const float uv = uptr[(long)l * D_INNER];
        const float zv = zptr[(long)l * (2 * D_INNER)];
        const float du = dv * uv;

        float yacc = 0.0f;
#pragma unroll
        for (int n = 0; n < D_STATE; ++n) {
            h[n] = __expf(dv * Arow[n]) * h[n] + du * sBC[n];
            yacc += h[n] * sBC[D_STATE + n];
        }
        float yv = yacc + uv * dp;
        yv *= zv / (1.0f + __expf(-zv));  // * silu(z)
        yptr[(long)l * D_INNER] = yv;
    }
}

// ---------------------------------------------------------------------------
extern "C" void kernel_launch(void* const* d_in, const int* in_sizes, int n_in,
                              void* d_out, int out_size, void* d_ws, size_t ws_size,
                              hipStream_t stream)
{
    const float* x_in  = (const float*)d_in[0];
    const float* ipw   = (const float*)d_in[1];
    const float* cw    = (const float*)d_in[2];
    const float* cb    = (const float*)d_in[3];
    const float* xpw   = (const float*)d_in[4];
    const float* dpw   = (const float*)d_in[5];
    const float* dpb   = (const float*)d_in[6];
    const float* alog  = (const float*)d_in[7];
    const float* dprm  = (const float*)d_in[8];
    const float* opw   = (const float*)d_in[9];
    float*       out   = (float*)d_out;

    // Workspace layout (fp32)
    float* ws    = (float*)d_ws;
    float* xbuf  = ws;                                      // 2048*768
    float* xz    = xbuf  + (long)MROWS * D_MODEL;           // 2048*3072
    float* u     = xz    + (long)MROWS * 2 * D_INNER;       // 2048*1536
    float* xdbl  = u     + (long)MROWS * D_INNER;           // 2048*80
    float* delta = xdbl  + (long)MROWS * XDBL_N;            // 2048*1536
    float* ybuf  = delta + (long)MROWS * D_INNER;           // 2048*1536
    float* dt64  = ybuf  + (long)MROWS * D_INNER;           // 2048*64
    float* dpw64 = dt64  + (long)MROWS * DT_PAD;            // 4*1536*64

    const dim3 blk(256);
    const int g_inproj  = (MROWS / 32) * (2 * D_INNER / 16) / 8; // 1536
    const int g_xproj   = (MROWS / 32) * (XDBL_N / 16) / 8;      // 40
    const int g_dtproj  = (MROWS / 32) * (D_INNER / 16) / 8;     // 768
    const int g_outproj = (MROWS / 32) * (D_MODEL / 16) / 8;     // 384
    const int g_conv    = (MROWS * D_INNER + 255) / 256;
    const int g_paddt   = (MROWS * DT_PAD + 255) / 256;
    const int g_paddpw  = (N_LAYERS * D_INNER * DT_PAD + 255) / 256;
    const int g_scan    = BATCH * (D_INNER / 256);               // 12

    // Pad dt_proj weights for all layers (K=48 -> 64) once per call.
    mamba_pad_dpw<<<g_paddpw, blk, 0, stream>>>(dpw, dpw64);

    for (int i = 0; i < N_LAYERS; ++i) {
        const float* layer_in  = (i == 0) ? x_in : xbuf;
        float*       layer_out = (i == N_LAYERS - 1) ? out : xbuf;

        // 1) xz = x @ in_proj_w^T   [2048, 3072]
        mamba_gemm_wmma_bf16<<<g_inproj, blk, 0, stream>>>(
            layer_in, D_MODEL, ipw + (long)i * 2 * D_INNER * D_MODEL, D_MODEL,
            xz, 2 * D_INNER, nullptr, MROWS, 2 * D_INNER, D_MODEL, EPI_NONE);

        // 2) u = silu(causal_dwconv(xz[:, :D_INNER]) + cb)
        mamba_conv_silu<<<g_conv, blk, 0, stream>>>(
            xz, cw + (long)i * D_INNER * D_CONV, cb + (long)i * D_INNER, u);

        // 3) x_dbl = u @ x_proj_w^T   [2048, 80]
        mamba_gemm_wmma_bf16<<<g_xproj, blk, 0, stream>>>(
            u, D_INNER, xpw + (long)i * XDBL_N * D_INNER, D_INNER,
            xdbl, XDBL_N, nullptr, MROWS, XDBL_N, D_INNER, EPI_NONE);

        // 3b) zero-pad dt columns 48..63
        mamba_pad_dt<<<g_paddt, blk, 0, stream>>>(xdbl, dt64);

        // 4) delta = softplus(dt @ dt_proj_w^T + dt_proj_b)   [2048, 1536]
        mamba_gemm_wmma_bf16<<<g_dtproj, blk, 0, stream>>>(
            dt64, DT_PAD, dpw64 + (long)i * D_INNER * DT_PAD, DT_PAD,
            delta, D_INNER, dpb + (long)i * D_INNER,
            MROWS, D_INNER, DT_PAD, EPI_BIAS_SOFTPLUS);

        // 5) selective scan fused with skip + silu(z) gating
        mamba_scan<<<g_scan, blk, 0, stream>>>(
            delta, u, xdbl, xz,
            alog + (long)i * D_INNER * D_STATE, dprm + (long)i * D_INNER, ybuf);

        // 6) out = y @ out_proj_w^T   [2048, 768]
        mamba_gemm_wmma_bf16<<<g_outproj, blk, 0, stream>>>(
            ybuf, D_INNER, opw + (long)i * D_MODEL * D_INNER, D_INNER,
            layer_out, D_MODEL, nullptr, MROWS, D_MODEL, D_INNER, EPI_NONE);
    }
}
